// ListenAttendSpell_83442624627377
// MI455X (gfx1250) — matrix operand order repaired and checked by reading, not compiled, not run on hardware
//
#include <hip/hip_runtime.h>
#include <hip/hip_bf16.h>

// ---------------------------------------------------------------------------
// Listen-Attend-Spell forward for MI455X (gfx1250, wave32, WMMA bf16->f32).
// Weights converted to bf16 once per call (~148MB -> resident in 192MB L2, so
// the 896 encoder steps + 128 decoder steps stream weights from L2, not HBM).
// Every GEMM is skinny [32 x K] x [K x N]: each wave computes a 32x32 tile
// with 4 v_wmma_f32_16x16x32_bf16 accumulators (2 A-frags x 2 B-frags).
// LSTM Wi/Wh GEMMs fused into one kernel (template<DUAL>).
// ---------------------------------------------------------------------------

typedef unsigned short u16;
typedef __attribute__((ext_vector_type(16))) __bf16 v16bf;
typedef __attribute__((ext_vector_type(8)))  float  v8f;

__device__ __forceinline__ __bf16 bf_bits(u16 u) {
  union { u16 u; __bf16 b; } c; c.u = u; return c.b;
}
__device__ __forceinline__ u16 f2bf(float f) {
  unsigned x = __float_as_uint(f);
  unsigned r = x + 0x7FFFu + ((x >> 16) & 1u);   // round-to-nearest-even
  return (u16)(r >> 16);
}
__device__ __forceinline__ float bf2f(u16 u) {
  return __uint_as_float(((unsigned)u) << 16);
}

// ---------------------------------------------------------------- conversion
__global__ void k_cvt_bf16(const float* __restrict__ src, u16* __restrict__ dst, long n) {
  long i = (long)blockIdx.x * blockDim.x + threadIdx.x;
  long s = (long)gridDim.x * blockDim.x;
  for (; i < n; i += s) dst[i] = f2bf(src[i]);
}

// ------------------------------------------------------------- WMMA fragments
// A fragment: 16x32 (MxK) bf16. lane&15 = row M; per ISA layout table:
//   lanes 0-15 : elems 0..7 -> K k0..k0+7,   elems 8..15 -> K k0+16..k0+23
//   lanes 16-31: elems 0..7 -> K k0+8..k0+15, elems 8..15 -> K k0+24..k0+31
// No M guard: all GEMMs in this model have M % 32 == 0.
__device__ __forceinline__ v16bf ld_afrag(const u16* __restrict__ A, long lda,
                                          int m0, int k0, int lane) {
  v16bf r;
  int m = m0 + (lane & 15);
  const u16* p = A + (long)m * lda + k0 + ((lane >> 4) << 3);
  union { uint4 v; u16 h[8]; } a, b;
  a.v = *(const uint4*)(p);
  b.v = *(const uint4*)(p + 16);
#pragma unroll
  for (int e = 0; e < 8; ++e) { r[e] = bf_bits(a.h[e]); r[8 + e] = bf_bits(b.h[e]); }
  return r;
}

// B fragment: 32x16 (KxN) bf16 from weight W stored [N,K] row-major
// (x @ W.T orientation: B[k,n] = W[n,k]; contiguous in K per lane).
//   lane&15 = column n; same K order as A fragment (elems 0..7 -> k0+8*(lane>>4)+e, 8..15 -> +16)
template<bool GUARD>
__device__ __forceinline__ v16bf ld_bfrag(const u16* __restrict__ W, long ldw,
                                          int n0, int k0, int lane, int N) {
  v16bf r;
  int n = n0 + (lane & 15);
  if (!GUARD || n < N) {
    const u16* p = W + (long)n * ldw + k0 + ((lane >> 4) << 3);
    union { uint4 v; u16 h[8]; } a, b;
    a.v = *(const uint4*)(p);
    b.v = *(const uint4*)(p + 16);
#pragma unroll
    for (int e = 0; e < 8; ++e) { r[e] = bf_bits(a.h[e]); r[8 + e] = bf_bits(b.h[e]); }
  } else {
#pragma unroll
    for (int e = 0; e < 16; ++e) r[e] = (__bf16)0.0f;
  }
  return r;
}

// K-loop core: accumulate a 32x32 tile. GUARD only for the N=5000 edge waves.
template<bool GUARD>
__device__ __forceinline__ void gemm_core(const u16* __restrict__ A, long lda,
                                          const u16* __restrict__ W, int K,
                                          int m0, int n0, int lane, int N,
                                          v8f acc[2][2]) {
  for (int k0 = 0; k0 < K; k0 += 32) {
    v16bf fa0 = ld_afrag(A, lda, m0,      k0, lane);
    v16bf fa1 = ld_afrag(A, lda, m0 + 16, k0, lane);
    v16bf fb0 = ld_bfrag<GUARD>(W, (long)K, n0,      k0, lane, N);
    v16bf fb1 = ld_bfrag<GUARD>(W, (long)K, n0 + 16, k0, lane, N);
    acc[0][0] = __builtin_amdgcn_wmma_f32_16x16x32_bf16(false, fa0, false, fb0, (short)0, acc[0][0], false, false);
    acc[1][0] = __builtin_amdgcn_wmma_f32_16x16x32_bf16(false, fa1, false, fb0, (short)0, acc[1][0], false, false);
    acc[0][1] = __builtin_amdgcn_wmma_f32_16x16x32_bf16(false, fa0, false, fb1, (short)0, acc[0][1], false, false);
    acc[1][1] = __builtin_amdgcn_wmma_f32_16x16x32_bf16(false, fa1, false, fb1, (short)0, acc[1][1], false, false);
  }
}

// ------------------------------------------------------------------ GEMM
// C[M,N] = A1[M,K1] @ W1[N,K1]^T (+ A2[M,K2] @ W2[N,K2]^T) + bias, activation.
// act: 0=none, 1=relu, 2=leaky(0.01). Optional f32 and/or bf16 outputs.
// Wave -> 32(M) x 32(N) tile; block of 8 waves -> 256 columns.
template<bool DUAL>
__global__ __launch_bounds__(256)
void k_gemm_t(const u16* __restrict__ A1, long lda1, int K1,
              const u16* __restrict__ A2, long lda2, int K2,
              const u16* __restrict__ W1, const u16* __restrict__ W2,
              const float* __restrict__ bias,
              float* __restrict__ Cf, long ldcf,
              u16* __restrict__ Cb, long ldcb,
              int N, int act) {
  int wave = threadIdx.x >> 5, lane = threadIdx.x & 31;
  int n0 = (blockIdx.x * 8 + wave) * 32;
  int m0 = blockIdx.y * 32;
  if (n0 >= N) return;                         // wave-uniform: EXEC stays full

  v8f acc[2][2];
#pragma unroll
  for (int i = 0; i < 2; ++i)
#pragma unroll
    for (int j = 0; j < 2; ++j)
#pragma unroll
      for (int r = 0; r < 8; ++r) acc[i][j][r] = 0.f;

  if (n0 + 32 <= N) {                          // interior: no guards in K loop
    gemm_core<false>(A1, lda1, W1, K1, m0, n0, lane, N, acc);
    if (DUAL) gemm_core<false>(A2, lda2, W2, K2, m0, n0, lane, N, acc);
  } else {                                     // edge tile (only the V=5000 head)
    gemm_core<true>(A1, lda1, W1, K1, m0, n0, lane, N, acc);
    if (DUAL) gemm_core<true>(A2, lda2, W2, K2, m0, n0, lane, N, acc);
  }

  int half = lane >> 4;
#pragma unroll
  for (int nt = 0; nt < 2; ++nt) {
    int nn = n0 + nt * 16 + (lane & 15);
    if (nn >= N) continue;
    float bv = bias ? bias[nn] : 0.0f;
#pragma unroll
    for (int r = 0; r < 8; ++r) {
      int m = m0 + r + 8 * half;               // C layout: VGPR r, lane half
      float v0 = acc[0][nt][r] + bv;
      float v1 = acc[1][nt][r] + bv;
      if (act == 1) { v0 = fmaxf(v0, 0.f); v1 = fmaxf(v1, 0.f); }
      else if (act == 2) {
        v0 = (v0 >= 0.f) ? v0 : 0.01f * v0;
        v1 = (v1 >= 0.f) ? v1 : 0.01f * v1;
      }
      if (Cf) { Cf[(long)m * ldcf + nn] = v0; Cf[(long)(m + 16) * ldcf + nn] = v1; }
      if (Cb) { Cb[(long)m * ldcb + nn] = f2bf(v0); Cb[(long)(m + 16) * ldcb + nn] = f2bf(v1); }
    }
  }
}

// ---------------------------------------------------------- LSTM pointwise
// z[B,4H] gate order i,f,g,o.  c updated in place (f32). h written bf16 to a
// compact [B,H] buffer (recurrence GEMM input) and optionally a strided copy
// (pyramidal layer output concat slot).
__global__ void k_lstm_pw(const float* __restrict__ z, float* __restrict__ c,
                          u16* __restrict__ hC, u16* __restrict__ hS, long ldhS, int H) {
  int b = blockIdx.x;
  const float* zb = z + (long)b * 4 * H;
  for (int e = threadIdx.x; e < H; e += blockDim.x) {
    float ig = zb[e], fg = zb[H + e], gg = zb[2 * H + e], og = zb[3 * H + e];
    float si = 1.f / (1.f + __expf(-ig));
    float sf = 1.f / (1.f + __expf(-fg));
    float so = 1.f / (1.f + __expf(-og));
    float cc = sf * c[(long)b * H + e] + si * tanhf(gg);
    c[(long)b * H + e] = cc;
    u16 hb = f2bf(so * tanhf(cc));
    hC[(long)b * H + e] = hb;
    if (hS) hS[(long)b * ldhS + e] = hb;
  }
}

// --------------------------------------------------- train-mode batchnorm+relu
__global__ void k_bn_relu(const float* __restrict__ Z, const float* __restrict__ g,
                          const float* __restrict__ be, u16* __restrict__ out,
                          int M, int N) {
  int col = blockIdx.x * blockDim.x + threadIdx.x;
  if (col >= N) return;
  float s = 0.f, s2 = 0.f;
  for (int i = 0; i < M; ++i) { float v = Z[(long)i * N + col]; s += v; s2 += v * v; }
  float mean = s / M;
  float var  = s2 / M - mean * mean;
  float inv  = rsqrtf(var + 1e-5f);
  float gc = g[col] * inv, bc = be[col] - mean * gc;
  for (int i = 0; i < M; ++i) {
    float v = Z[(long)i * N + col] * gc + bc;
    out[(long)i * N + col] = f2bf(fmaxf(v, 0.f));
  }
}

// ---------------------------------------------------------------- attention
// scores[b,t] = <hs[b,t,:], s[b,:]>, softmax over t (Tc=64), ctx = att @ hs.
__global__ __launch_bounds__(256)
void k_attn(const u16* __restrict__ hs, const u16* __restrict__ s, long lds_,
            u16* __restrict__ ctx, long ldctx, int Tc, int E) {
  __shared__ float sc[64];
  __shared__ float att[64];
  int b = blockIdx.x;
  int wave = threadIdx.x >> 5, lane = threadIdx.x & 31;
  const u16* hb = hs + (long)b * Tc * E;
  const u16* sb = s + (long)b * lds_;
  for (int j = 0; j < 8; ++j) {                    // 8 waves x 8 t-values = 64
    int t = wave * 8 + j;
    float p = 0.f;
    const u16* row = hb + (long)t * E;
    for (int e = lane; e < E; e += 32) p += bf2f(row[e]) * bf2f(sb[e]);
#pragma unroll
    for (int m = 16; m >= 1; m >>= 1) p += __shfl_xor(p, m, 32);
    if (lane == 0) sc[t] = p;
  }
  __syncthreads();
  float mx = -3.4e38f;
  for (int t = 0; t < Tc; ++t) mx = fmaxf(mx, sc[t]);
  if ((int)threadIdx.x < Tc) att[threadIdx.x] = __expf(sc[threadIdx.x] - mx);
  __syncthreads();
  float sum = 0.f;
  for (int t = 0; t < Tc; ++t) sum += att[t];
  float inv = 1.f / sum;
  for (int e = threadIdx.x; e < E; e += blockDim.x) {
    float a = 0.f;
    for (int t = 0; t < Tc; ++t) a += att[t] * bf2f(hb[(long)t * E + e]);
    ctx[(long)b * ldctx + e] = f2bf(a * inv);
  }
}

// --------------------------------------- teacher-force select + embedding gather
__global__ void k_sel_embed(const int* __restrict__ chars, const int* __restrict__ mask,
                            const int* __restrict__ pred, int l, int L,
                            const u16* __restrict__ emb, u16* __restrict__ out, int E) {
  int b = blockIdx.x;
  int ch;
  if (pred == nullptr) ch = chars[b * L + l];
  else { int m = mask[b * L + l]; ch = m ? chars[b * L + l] : pred[b]; }
  const u16* row = emb + (long)ch * E;
  for (int e = threadIdx.x; e < E; e += blockDim.x) out[(long)b * E + e] = row[e];
}

// ---------------------------------------------------------------- row argmax
__global__ __launch_bounds__(256)
void k_argmax(const float* __restrict__ logits, long ld, int V, int* __restrict__ pred) {
  __shared__ float bv[256];
  __shared__ int   bi[256];
  int b = blockIdx.x;
  const float* row = logits + (long)b * ld;
  float best = -3.4e38f; int bidx = 0;
  for (int v = threadIdx.x; v < V; v += blockDim.x) {
    float x = row[v];
    if (x > best || (x == best && v < bidx)) { best = x; bidx = v; }
  }
  bv[threadIdx.x] = best; bi[threadIdx.x] = bidx;
  __syncthreads();
  for (int s = 128; s > 0; s >>= 1) {
    if ((int)threadIdx.x < s) {
      float ov = bv[threadIdx.x + s]; int oi = bi[threadIdx.x + s];
      if (ov > bv[threadIdx.x] || (ov == bv[threadIdx.x] && oi < bi[threadIdx.x])) {
        bv[threadIdx.x] = ov; bi[threadIdx.x] = oi;
      }
    }
    __syncthreads();
  }
  if (threadIdx.x == 0) pred[b] = bi[0];
}

// ===========================================================================
extern "C" void kernel_launch(void* const* d_in, const int* in_sizes, int n_in,
                              void* d_out, int out_size, void* d_ws, size_t ws_size,
                              hipStream_t stream) {
  (void)in_sizes; (void)n_in; (void)out_size; (void)ws_size;
  const int B = 32, E = 1024, V = 5000, L = 128, H = 512;

  // ---- input pointer map (setup_inputs() insertion order, leaves in order) ----
  int ii = 0;
  const float* feats = (const float*)d_in[ii++];   // [32,512,512]
  const int*   chars = (const int*)d_in[ii++];     // [32,128]
  const int*   tfm   = (const int*)d_in[ii++];     // [32,128]
  const float* encP[3][2][3];                      // [layer][fwd/bwd][Wi,Wh,b]
  for (int l = 0; l < 3; ++l)
    for (int d = 0; d < 2; ++d)
      for (int k = 0; k < 3; ++k) encP[l][d][k] = (const float*)d_in[ii++];
  const float* fmH[2][4];                          // feat_mlp hidden: W,b,g,be
  for (int h = 0; h < 2; ++h) for (int k = 0; k < 4; ++k) fmH[h][k] = (const float*)d_in[ii++];
  const float* fmoW = (const float*)d_in[ii++];
  const float* fmob = (const float*)d_in[ii++];
  const float* embF = (const float*)d_in[ii++];    // [5000,1024]
  const float* cmH[2][4];
  for (int h = 0; h < 2; ++h) for (int k = 0; k < 4; ++k) cmH[h][k] = (const float*)d_in[ii++];
  const float* cmoW = (const float*)d_in[ii++];
  const float* cmob = (const float*)d_in[ii++];
  const float* dcP[3][3];                          // dec LSTMs: Wi,Wh,b
  for (int s = 0; s < 3; ++s) for (int k = 0; k < 3; ++k) dcP[s][k] = (const float*)d_in[ii++];
  const float* l1W = (const float*)d_in[ii++];
  const float* l1b = (const float*)d_in[ii++];
  const float* l2W = (const float*)d_in[ii++];
  const float* l2b = (const float*)d_in[ii++];

  // ---- workspace bump allocator ----
  char* wsp = (char*)d_ws;
  size_t off = 0;
  auto alloc = [&](size_t bytes) -> void* {
    void* p = wsp + off;
    off = (off + bytes + 255) & ~(size_t)255;
    return p;
  };
  auto cvt = [&](const float* src, long n) -> u16* {
    u16* d = (u16*)alloc((size_t)n * sizeof(u16));
    long bl = (n + 255) / 256; if (bl > 8192) bl = 8192;
    k_cvt_bf16<<<dim3((unsigned)bl), 256, 0, stream>>>(src, d, n);
    return d;
  };

  // ---- bf16 weight mirror (~148MB: resident in the 192MB L2 across steps) ----
  u16 *encWi[3][2], *encWh[3][2];
  for (int l = 0; l < 3; ++l)
    for (int d = 0; d < 2; ++d) {
      long ki = (l == 0) ? 1024 : 2048;
      encWi[l][d] = cvt(encP[l][d][0], 2048L * ki);
      encWh[l][d] = cvt(encP[l][d][1], 2048L * 512);
    }
  u16* fmWb[2]; for (int h = 0; h < 2; ++h) fmWb[h] = cvt(fmH[h][0], 1024L * 1024);
  u16* fmoWb = cvt(fmoW, 1024L * 1024);
  u16* embB  = cvt(embF, (long)V * E);
  u16* cmWb[2]; for (int h = 0; h < 2; ++h) cmWb[h] = cvt(cmH[h][0], 1024L * 1024);
  u16* cmoWb = cvt(cmoW, 1024L * 1024);
  u16 *dcWi[3], *dcWh[3];
  for (int s = 0; s < 3; ++s) {
    long ki = (s == 0) ? 2048 : 1024;
    dcWi[s] = cvt(dcP[s][0], 4096L * ki);
    dcWh[s] = cvt(dcP[s][1], 4096L * 1024);
  }
  u16* l1Wb = cvt(l1W, 1024L * 1024);
  u16* l2Wb = cvt(l2W, (long)V * E);
  u16* featsB = cvt(feats, 32L * 512 * 512);       // [B,256,1024] view

  // ---- activation buffers ----
  u16* y[3];
  y[0] = (u16*)alloc((size_t)B * 256 * 1024 * 2);
  y[1] = (u16*)alloc((size_t)B * 128 * 1024 * 2);
  y[2] = (u16*)alloc((size_t)B * 64  * 1024 * 2);
  float* zenc = (float*)alloc((size_t)B * 2048 * 4);
  u16*   henc = (u16*)alloc((size_t)B * H * 2);
  float* cenc = (float*)alloc((size_t)B * H * 4);
  float* zmlp = (float*)alloc((size_t)B * 64 * 1024 * 4);
  u16*   amlp = (u16*)alloc((size_t)B * 64 * 1024 * 2);
  u16*   hsB  = (u16*)alloc((size_t)B * 64 * 1024 * 2);
  u16*   sIn  = (u16*)alloc((size_t)B * E * 2);
  float* zc   = (float*)alloc((size_t)B * E * 4);
  u16*   ac   = (u16*)alloc((size_t)B * E * 2);
  u16*   xx   = (u16*)alloc((size_t)B * 2 * E * 2);    // [s | ctx]
  float* zdec = (float*)alloc((size_t)B * 4096 * 4);
  u16*   hdec[3]; float* cdec[3];
  for (int s = 0; s < 3; ++s) {
    hdec[s] = (u16*)alloc((size_t)B * E * 2);
    cdec[s] = (float*)alloc((size_t)B * E * 4);
  }
  u16* zl   = (u16*)alloc((size_t)B * E * 2);
  int* pred = (int*)alloc((size_t)B * 4);

  // single-input GEMM
  auto gemm1 = [&](const u16* A1, long lda1, int K1,
                   const u16* W1, const float* bias,
                   float* Cf, long ldcf, u16* Cb, long ldcb,
                   int M, int N, int act) {
    dim3 grid((unsigned)((N + 255) / 256), (unsigned)((M + 31) / 32));
    k_gemm_t<false><<<grid, 256, 0, stream>>>(A1, lda1, K1, nullptr, 0, 0,
                                              W1, nullptr, bias, Cf, ldcf, Cb, ldcb, N, act);
  };
  // fused dual-input GEMM (LSTM Wi + Wh)
  auto gemm2 = [&](const u16* A1, long lda1, int K1,
                   const u16* A2, long lda2, int K2,
                   const u16* W1, const u16* W2, const float* bias,
                   float* Cf, long ldcf, int M, int N) {
    dim3 grid((unsigned)((N + 255) / 256), (unsigned)((M + 31) / 32));
    k_gemm_t<true><<<grid, 256, 0, stream>>>(A1, lda1, K1, A2, lda2, K2,
                                             W1, W2, bias, Cf, ldcf, nullptr, 0, N, 0);
  };

  // ================= Listener: 3 pyramidal BiLSTM layers =================
  const u16* xin = featsB;
  int Tcur = 256, insz = 1024;                     // [B,256,1024] after pair-concat view
  for (int layer = 0; layer < 3; ++layer) {
    u16* yout = y[layer];                          // [B,Tcur,1024] = concat(fwd,bwd)
    for (int d = 0; d < 2; ++d) {
      hipMemsetAsync(henc, 0, (size_t)B * H * 2, stream);
      hipMemsetAsync(cenc, 0, (size_t)B * H * 4, stream);
      for (int s = 0; s < Tcur; ++s) {
        int t = d ? (Tcur - 1 - s) : s;
        const u16* At = xin + (long)t * insz;      // per-sample row stride Tcur*insz
        gemm2(At, (long)Tcur * insz, insz, henc, H, H,
              encWi[layer][d], encWh[layer][d], encP[layer][d][2],
              zenc, 4 * H, B, 4 * H);
        k_lstm_pw<<<B, 256, 0, stream>>>(zenc, cenc, henc,
              yout + (long)t * (2 * H) + d * H, (long)Tcur * 2 * H, H);
      }
    }
    xin = yout;
    insz = 2048;                                   // pair-concat decimation (view)
    Tcur >>= 1;
  }

  // ================= feat MLP: [2048,1024] -> hs bf16 =================
  {
    int Mm = B * 64;
    gemm1(y[2], 1024, 1024, fmWb[0], fmH[0][1], zmlp, 1024, nullptr, 0, Mm, 1024, 0);
    k_bn_relu<<<4, 256, 0, stream>>>(zmlp, fmH[0][2], fmH[0][3], amlp, Mm, 1024);
    gemm1(amlp, 1024, 1024, fmWb[1], fmH[1][1], zmlp, 1024, nullptr, 0, Mm, 1024, 0);
    k_bn_relu<<<4, 256, 0, stream>>>(zmlp, fmH[1][2], fmH[1][3], amlp, Mm, 1024);
    gemm1(amlp, 1024, 1024, fmoWb, fmob, nullptr, 0, hsB, 1024, Mm, 1024, 0);
  }

  // ================= Speller: 128 decode steps =================
  for (int s = 0; s < 3; ++s) {
    hipMemsetAsync(hdec[s], 0, (size_t)B * E * 2, stream);
    hipMemsetAsync(cdec[s], 0, (size_t)B * E * 4, stream);
  }
  for (int l = 0; l < L; ++l) {
    k_sel_embed<<<B, 256, 0, stream>>>(chars, tfm, (l == 0) ? nullptr : pred,
                                       l, L, embB, sIn, E);
    // char_mlp -> s, written into xx[:, :E]
    gemm1(sIn, E, E, cmWb[0], cmH[0][1], zc, E, nullptr, 0, B, E, 0);
    k_bn_relu<<<4, 256, 0, stream>>>(zc, cmH[0][2], cmH[0][3], ac, B, E);
    gemm1(ac, E, E, cmWb[1], cmH[1][1], zc, E, nullptr, 0, B, E, 0);
    k_bn_relu<<<4, 256, 0, stream>>>(zc, cmH[1][2], cmH[1][3], ac, B, E);
    gemm1(ac, E, E, cmoWb, cmob, nullptr, 0, xx, 2 * E, B, E, 0);
    // attention -> ctx into xx[:, E:2E]
    k_attn<<<B, 256, 0, stream>>>(hsB, xx, 2 * E, xx + E, 2 * E, 64, E);
    // 3 decoder LSTM cells (Wi/Wh GEMMs fused)
    const u16* xd = xx; long ldx = 2 * E; int kx = 2 * E;
    for (int s = 0; s < 3; ++s) {
      gemm2(xd, ldx, kx, hdec[s], E, E, dcWi[s], dcWh[s], dcP[s][2],
            zdec, 4 * E, B, 4 * E);
      k_lstm_pw<<<B, 256, 0, stream>>>(zdec, cdec[s], hdec[s], nullptr, 0, E);
      xd = hdec[s]; ldx = E; kx = E;
    }
    // output head: linear -> leaky relu -> linear -> logits[l] in d_out
    gemm1(hdec[2], E, E, l1Wb, l1b, nullptr, 0, zl, E, B, E, 2);
    float* logit = (float*)d_out + (long)l * B * V;
    gemm1(zl, E, E, l2Wb, l2b, logit, V, nullptr, 0, B, V, 0);
    if (l + 1 < L)
      k_argmax<<<B, 256, 0, stream>>>(logit, V, V, pred);
  }
}
